// BERTLegalWithLawEmb_90615220011636
// MI455X (gfx1250) — matrix operand order repaired and checked by reading, not compiled, not run on hardware
//
#include <hip/hip_runtime.h>
#include <hip/hip_bf16.h>
#include <math.h>

constexpr int DIM   = 768;
constexpr int LAWC  = 103;
constexpr int ACCUC = 119;
constexpr int TERMC = 11;
constexpr int LSEQ  = 512;
constexpr int BATCH = 2;
constexpr int SEQ   = 512;

typedef unsigned short ushort_t;
typedef __attribute__((ext_vector_type(16))) __bf16 v16bf;
typedef __attribute__((ext_vector_type(8)))  float  v8f;

union FragBF { uint4 u[2]; v16bf v; };

#ifndef __has_builtin
#define __has_builtin(x) 0
#endif

#if defined(__gfx1250__) && \
    __has_builtin(__builtin_amdgcn_global_load_async_to_lds_b128) && \
    __has_builtin(__builtin_amdgcn_s_wait_asynccnt)
#define USE_ASYNC_LDS 1
// Types matching the builtin's expected parameters exactly (from hipcc diagnostic):
//   param1: '__attribute__((__vector_size__(4*sizeof(int)))) int __device__ *'  (AS1 global)
//   param2: LDS pointer (AS3)
typedef int v4i_t __attribute__((vector_size(16)));
typedef __attribute__((address_space(1))) v4i_t* as1_v4i_t;
typedef __attribute__((address_space(3))) v4i_t* as3_v4i_t;
#else
#define USE_ASYNC_LDS 0
#endif

__device__ __forceinline__ ushort_t f32_to_bf16(float f) {
    unsigned u = __float_as_uint(f);
    unsigned r = u + 0x7FFFu + ((u >> 16) & 1u);   // round-to-nearest-even
    return (ushort_t)(r >> 16);
}

// ---------------- conversion kernels ----------------
__global__ void k_cvt_bf16(const float* __restrict__ src, ushort_t* __restrict__ dst, int n) {
    int i = blockIdx.x * blockDim.x + threadIdx.x;
    if (i < n) dst[i] = f32_to_bf16(src[i]);
}

// dst[c*R + r] = src[r*C + c]  (store transposed, bf16)
__global__ void k_cvt_bf16_T(const float* __restrict__ src, ushort_t* __restrict__ dst, int R, int C) {
    int i = blockIdx.x * blockDim.x + threadIdx.x;
    if (i < R * C) {
        int r = i / C, c = i % C;
        dst[c * R + r] = f32_to_bf16(src[i]);
    }
}

// ---------------- small prep: wvd = Wv @ w_dprob, wvd[DIM] = bv . w_dprob ----------------
__global__ void k_prep(const float* __restrict__ Wv, const float* __restrict__ bv,
                       const float* __restrict__ w_dprob, float* __restrict__ wvd) {
    for (int i = threadIdx.x; i < DIM; i += blockDim.x) {
        const float* r = Wv + (size_t)i * DIM;
        float s = 0.f;
        for (int j = 0; j < DIM; j++) s += r[j] * w_dprob[j];
        wvd[i] = s;
    }
    if (threadIdx.x == 0) {
        float s = 0.f;
        for (int j = 0; j < DIM; j++) s += bv[j] * w_dprob[j];
        wvd[DIM] = s;
    }
}

// ---------------- cls heads: law0 (ws), accu/term (d_out) ----------------
__global__ void k_cls(const float* __restrict__ output,
                      const float* __restrict__ W_law,  const float* __restrict__ b_law,
                      const float* __restrict__ W_accu, const float* __restrict__ b_accu,
                      const float* __restrict__ W_term, const float* __restrict__ b_term,
                      float* __restrict__ law0, float* __restrict__ d_out) {
    int tot = BATCH * (LAWC + ACCUC + TERMC);
    for (int t = blockIdx.x * blockDim.x + threadIdx.x; t < tot; t += blockDim.x * gridDim.x) {
        if (t < BATCH * LAWC) {
            int f = t / LAWC, c = t % LAWC;
            const float* cls = output + (size_t)f * SEQ * DIM;
            float v = b_law[c];
            for (int d = 0; d < DIM; d++) v += cls[d] * W_law[d * LAWC + c];
            law0[t] = v;
        } else if (t < BATCH * (LAWC + ACCUC)) {
            int j = t - BATCH * LAWC;
            int f = j / ACCUC, c = j % ACCUC;
            const float* cls = output + (size_t)f * SEQ * DIM;
            float v = b_accu[c];
            for (int d = 0; d < DIM; d++) v += cls[d] * W_accu[d * ACCUC + c];
            d_out[2 * LAWC + j] = v;
        } else {
            int j = t - BATCH * (LAWC + ACCUC);
            int f = j / TERMC, c = j % TERMC;
            const float* cls = output + (size_t)f * SEQ * DIM;
            float v = b_term[c];
            for (int d = 0; d < DIM; d++) v += cls[d] * W_term[d * TERMC + c];
            d_out[2 * LAWC + 2 * ACCUC + j] = v;
        }
    }
}

// ---------------- generic row-dot: out[row] = A[row,:] . vec (+ *addp) ----------------
__global__ void k_rowdot(const float* __restrict__ A, const float* __restrict__ vec,
                         const float* __restrict__ addp, float* __restrict__ out, int R) {
    int wave = threadIdx.x >> 5, lane = threadIdx.x & 31;
    int row = blockIdx.x * 8 + wave;
    if (row >= R) return;
    const float* a = A + (size_t)row * DIM;
    float s = 0.f;
    for (int j = lane; j < DIM; j += 32) s += a[j] * vec[j];
    for (int off = 16; off > 0; off >>= 1) s += __shfl_xor(s, off, 32);
    if (lane == 0) out[row] = s + (addp ? addp[0] : 0.0f);
}

// ---------------- WMMA GEMM: C[M,N] = A[M,K] @ Bt[N,K]^T + bias ----------------
__global__ __launch_bounds__(256)
void k_gemm_abT(const ushort_t* __restrict__ A, const ushort_t* __restrict__ Bt,
                const float* __restrict__ bias, float* __restrict__ Cf,
                ushort_t* __restrict__ Cb, int M, int N, int K) {
    int wave = threadIdx.x >> 5, lane = threadIdx.x & 31;
    int hf = lane >> 4, ln = lane & 15;
    int n0 = blockIdx.x * 128 + wave * 16;
    int m0 = blockIdx.y * 16;
    const ushort_t* ap = A  + (size_t)(m0 + ln) * K;
    const ushort_t* bp = Bt + (size_t)(n0 + ln) * K;
    v8f acc = {};
    for (int k0 = 0; k0 < K; k0 += 32) {
        FragBF a, b;
        // A 16x32 bf16: lanes 0-15 -> K {k0..k0+7, k0+16..k0+23}; lanes 16-31 -> +8
        a.u[0] = *(const uint4*)(ap + k0 + hf * 8);
        a.u[1] = *(const uint4*)(ap + k0 + hf * 8 + 16);
        // B 32x16 bf16: lanes 0-15 -> K {k0..k0+7, k0+16..k0+23}; lanes 16-31 -> +8
        b.u[0] = *(const uint4*)(bp + k0 + hf * 8);
        b.u[1] = *(const uint4*)(bp + k0 + hf * 8 + 16);
        acc = __builtin_amdgcn_wmma_f32_16x16x32_bf16(false, a.v, false, b.v,
                                                      (short)0, acc, false, false);
    }
    float bsv = bias ? bias[n0 + ln] : 0.0f;
#pragma unroll
    for (int i = 0; i < 8; i++) {
        int m = m0 + hf * 8 + i;
        int n = n0 + ln;
        float v = acc[i] + bsv;
        if (Cf) Cf[(size_t)m * N + n] = v;
        if (Cb) Cb[(size_t)m * N + n] = f32_to_bf16(v);
    }
}

// ---------------- fused attention: scores -> online softmax -> dot with vproj ----------------
__global__ __launch_bounds__(256, 1)
void k_attention(const ushort_t* __restrict__ q2, const float* __restrict__ c_fs,
                 const ushort_t* __restrict__ lawe, const float* __restrict__ law_masks,
                 const float* __restrict__ vproj, const float* __restrict__ b_dprob,
                 float* __restrict__ laws2) {
#if USE_ASYNC_LDS
    __shared__ __align__(16) ushort_t tile[2][16 * DIM];   // 2 x 24 KB double buffer
#else
    __shared__ __align__(16) ushort_t tile[1][16 * DIM];   // 24 KB single buffer
#endif
    int b   = blockIdx.x;
    int f   = b / (LAWC * 4);
    int rem = b % (LAWC * 4);
    int l   = rem >> 2;
    int st  = rem & 3;
    int wave = threadIdx.x >> 5, lane = threadIdx.x & 31;
    int hf = lane >> 4, ln = lane & 15;
    int s_base = st * 128 + wave * 16;

    // Load this wave's 24 A-fragments (its 16 query rows, full K=768) into registers.
    const ushort_t* ap = q2 + (size_t)(f * SEQ + s_base + ln) * DIM;
    v16bf afr[24];
#pragma unroll
    for (int kk = 0; kk < 24; kk++) {
        FragBF t;
        t.u[0] = *(const uint4*)(ap + kk * 32 + hf * 8);
        t.u[1] = *(const uint4*)(ap + kk * 32 + hf * 8 + 16);
        afr[kk] = t.v;
    }
    float cv[8], mrun[8], drun[8], nrun[8];
#pragma unroll
    for (int i = 0; i < 8; i++) {
        cv[i]   = c_fs[f * SEQ + s_base + hf * 8 + i];
        mrun[i] = -3.0e38f; drun[i] = 0.f; nrun[i] = 0.f;
    }
    const float scale = 0.03608439182435161f;   // 1/sqrt(768)
    const ushort_t* lsrc = lawe + (size_t)l * LSEQ * DIM;

    // --- one 16-key tile: 24 WMMAs + online-softmax merge ---
    auto compute_tile = [&](const ushort_t* tb, int r0) {
        v8f acc = {};
#pragma unroll
        for (int kk = 0; kk < 24; kk++) {
            FragBF bf;
            const ushort_t* bp = tb + ln * DIM + kk * 32 + hf * 8;
            bf.u[0] = *(const uint4*)bp;
            bf.u[1] = *(const uint4*)(bp + 16);
            acc = __builtin_amdgcn_wmma_f32_16x16x32_bf16(false, afr[kk], false, bf.v,
                                                          (short)0, acc, false, false);
        }
        float maskv = law_masks[l * LSEQ + r0 + ln];
        float vp    = vproj[l * LSEQ + r0 + ln];
#pragma unroll
        for (int i = 0; i < 8; i++) {
            float sc = (acc[i] + cv[i]) * scale + maskv;
            float tmax = sc;
            tmax = fmaxf(tmax, __shfl_xor(tmax, 1, 32));
            tmax = fmaxf(tmax, __shfl_xor(tmax, 2, 32));
            tmax = fmaxf(tmax, __shfl_xor(tmax, 4, 32));
            tmax = fmaxf(tmax, __shfl_xor(tmax, 8, 32));
            float e  = __expf(sc - tmax);
            float se = e, sv = e * vp;
            se += __shfl_xor(se, 1, 32); sv += __shfl_xor(sv, 1, 32);
            se += __shfl_xor(se, 2, 32); sv += __shfl_xor(sv, 2, 32);
            se += __shfl_xor(se, 4, 32); sv += __shfl_xor(sv, 4, 32);
            se += __shfl_xor(se, 8, 32); sv += __shfl_xor(sv, 8, 32);
            float nm = fmaxf(mrun[i], tmax);
            float g1 = __expf(mrun[i] - nm), g2 = __expf(tmax - nm);
            drun[i] = drun[i] * g1 + se * g2;
            nrun[i] = nrun[i] * g1 + sv * g2;
            mrun[i] = nm;
        }
    };

#if USE_ASYNC_LDS
    // --- CDNA5 async memory->LDS path (ASYNCcnt), double buffered ---
    auto stage = [&](int rt, int buf) {
        const uint4* src = (const uint4*)(lsrc + (size_t)rt * 16 * DIM);
        uint4* dst = (uint4*)tile[buf];
        for (int idx = threadIdx.x; idx < 16 * DIM / 8; idx += 256) {
            __builtin_amdgcn_global_load_async_to_lds_b128(
                (as1_v4i_t)(unsigned long long)(uintptr_t)(src + idx),
                (as3_v4i_t)(unsigned)(uintptr_t)(dst + idx),
                0, 0);
        }
    };
    stage(0, 0);
    __builtin_amdgcn_s_wait_asynccnt(0);
    __syncthreads();
#pragma unroll 1
    for (int rt = 0; rt < 32; rt++) {
        if (rt + 1 < 32) stage(rt + 1, (rt + 1) & 1);   // overlap copy with WMMA below
        compute_tile(tile[rt & 1], rt * 16);
        __builtin_amdgcn_s_wait_asynccnt(0);
        __syncthreads();
    }
#else
    // --- synchronous fallback (VGPR-staged copy) ---
#pragma unroll 1
    for (int rt = 0; rt < 32; rt++) {
        int r0 = rt * 16;
        __syncthreads();
        {
            const uint4* src = (const uint4*)(lsrc + (size_t)r0 * DIM);
            uint4* dst = (uint4*)tile[0];
            for (int idx = threadIdx.x; idx < 16 * DIM / 8; idx += 256) dst[idx] = src[idx];
        }
        __syncthreads();
        if (rt + 1 < 32)
            __builtin_prefetch((const char*)(lsrc + (size_t)(r0 + 16) * DIM) + threadIdx.x * 96, 0, 0);
        compute_tile(tile[0], r0);
    }
#endif

    if (ln == 0) {   // lanes 0 (rows 0-7) and 16 (rows 8-15) hold the reduced values
        float bd = b_dprob[0];
        size_t base = ((size_t)(f * LAWC + l)) * SEQ + s_base + hf * 8;
#pragma unroll
        for (int i = 0; i < 8; i++) laws2[base + i] = nrun[i] / drun[i] + bd;
    }
}

// ---------------- epilogue: scores2, law, argmax, law_language_pred ----------------
__global__ void k_final(const float* __restrict__ laws2, const float* __restrict__ w_lawprob,
                        const float* __restrict__ b_lawprob, const float* __restrict__ law0,
                        const float* __restrict__ W_rulelaw, const float* __restrict__ b_rulelaw,
                        const float* __restrict__ W_factlaw, const float* __restrict__ b_factlaw,
                        const float* __restrict__ laws_table, const float* __restrict__ W_law,
                        const float* __restrict__ b_law, float* __restrict__ d_out) {
    __shared__ float s2[BATCH * LAWC];
    __shared__ float lf[BATCH * LAWC];
    __shared__ int   lawno[BATCH];
    int t = threadIdx.x;
    if (t < BATCH * LAWC) {
        const float* row = laws2 + (size_t)t * SEQ;
        float s = b_lawprob[0];
        for (int j = 0; j < SEQ; j++) s += row[j] * w_lawprob[j];
        s2[t] = s;
    }
    __syncthreads();
    if (t < BATCH * LAWC) {
        int f = t / LAWC, c = t % LAWC;
        float v = b_rulelaw[c] + b_factlaw[c];
        for (int l = 0; l < LAWC; l++)
            v += law0[f * LAWC + l] * W_rulelaw[l * LAWC + c]
               + s2[f * LAWC + l]   * W_factlaw[l * LAWC + c];
        lf[t] = v;
        d_out[t] = v;                                    // law  [0, 206)
    }
    __syncthreads();
    if (t < BATCH) {
        int best = 0; float bv = lf[t * LAWC];
        for (int c = 1; c < LAWC; c++) { float v = lf[t * LAWC + c]; if (v > bv) { bv = v; best = c; } }
        lawno[t] = best;
        d_out[2 * LAWC + 2 * ACCUC + 2 * TERMC + 2 * LAWC + t] = (float)best;   // law_no [672, 674)
    }
    __syncthreads();
    if (t < BATCH * LAWC) {
        int f = t / LAWC, c = t % LAWC;
        const float* row = laws_table + (size_t)lawno[f] * DIM;
        float v = b_law[c];
        for (int d = 0; d < DIM; d++) v += row[d] * W_law[d * LAWC + c];
        d_out[2 * LAWC + 2 * ACCUC + 2 * TERMC + t] = v;                        // llp [466, 672)
    }
}

extern "C" void kernel_launch(void* const* d_in, const int* in_sizes, int n_in,
                              void* d_out_v, int out_size, void* d_ws, size_t ws_size,
                              hipStream_t stream) {
    (void)in_sizes; (void)n_in; (void)out_size; (void)ws_size;
    const float* output     = (const float*)d_in[0];
    const float* fact_emb   = (const float*)d_in[1];
    const float* law_embs   = (const float*)d_in[2];
    const float* law_masks  = (const float*)d_in[3];
    const float* laws_table = (const float*)d_in[4];
    const float* Wq = (const float*)d_in[5];  const float* bq = (const float*)d_in[6];
    const float* Wk = (const float*)d_in[7];  const float* bk = (const float*)d_in[8];
    const float* Wv = (const float*)d_in[9];  const float* bv = (const float*)d_in[10];
    const float* w_dprob   = (const float*)d_in[11]; const float* b_dprob   = (const float*)d_in[12];
    const float* w_lawprob = (const float*)d_in[13]; const float* b_lawprob = (const float*)d_in[14];
    const float* W_rulelaw = (const float*)d_in[15]; const float* b_rulelaw = (const float*)d_in[16];
    const float* W_factlaw = (const float*)d_in[17]; const float* b_factlaw = (const float*)d_in[18];
    const float* W_law  = (const float*)d_in[19]; const float* b_law  = (const float*)d_in[20];
    const float* W_accu = (const float*)d_in[21]; const float* b_accu = (const float*)d_in[22];
    const float* W_term = (const float*)d_in[23]; const float* b_term = (const float*)d_in[24];
    float* d_out = (float*)d_out_v;

    char* ws = (char*)d_ws;
    size_t off = 0;
    auto alloc = [&](size_t bytes) -> char* {
        char* p = ws + off; off = (off + bytes + 255) & ~(size_t)255; return p;
    };
    ushort_t* fact_bf   = (ushort_t*)alloc((size_t)BATCH * SEQ * DIM * 2);
    ushort_t* WqT_bf    = (ushort_t*)alloc((size_t)DIM * DIM * 2);
    ushort_t* Wk_bf     = (ushort_t*)alloc((size_t)DIM * DIM * 2);
    ushort_t* lawe_bf   = (ushort_t*)alloc((size_t)LAWC * LSEQ * DIM * 2);
    float*    factsq_f  = (float*)   alloc((size_t)BATCH * SEQ * DIM * 4);
    ushort_t* factsq_bf = (ushort_t*)alloc((size_t)BATCH * SEQ * DIM * 2);
    ushort_t* q2_bf     = (ushort_t*)alloc((size_t)BATCH * SEQ * DIM * 2);
    float*    c_fs      = (float*)   alloc((size_t)BATCH * SEQ * 4);
    float*    wvd       = (float*)   alloc((size_t)(DIM + 1) * 4);
    float*    vproj     = (float*)   alloc((size_t)LAWC * LSEQ * 4);
    float*    law0      = (float*)   alloc((size_t)BATCH * LAWC * 4);
    float*    laws2     = (float*)   alloc((size_t)BATCH * LAWC * SEQ * 4);

    int n1 = BATCH * SEQ * DIM;
    int n2 = LAWC * LSEQ * DIM;
    k_cvt_bf16  <<<(n1 + 255) / 256, 256, 0, stream>>>(fact_emb, fact_bf, n1);
    k_cvt_bf16_T<<<(DIM * DIM + 255) / 256, 256, 0, stream>>>(Wq, WqT_bf, DIM, DIM);
    k_cvt_bf16  <<<(DIM * DIM + 255) / 256, 256, 0, stream>>>(Wk, Wk_bf, DIM * DIM);
    k_cvt_bf16  <<<(n2 + 255) / 256, 256, 0, stream>>>(law_embs, lawe_bf, n2);
    k_prep<<<1, 256, 0, stream>>>(Wv, bv, w_dprob, wvd);
    k_cls <<<2, 256, 0, stream>>>(output, W_law, b_law, W_accu, b_accu, W_term, b_term, law0, d_out);

    dim3 g1(DIM / 128, BATCH * SEQ / 16);
    // facts_q = fact_emb @ Wq + bq (f32 + bf16 copies)
    k_gemm_abT<<<g1, 256, 0, stream>>>(fact_bf, WqT_bf, bq, factsq_f, factsq_bf, BATCH * SEQ, DIM, DIM);
    // c[s] = facts_q[s] . bk
    k_rowdot<<<(BATCH * SEQ) / 8, 256, 0, stream>>>(factsq_f, bk, nullptr, c_fs, BATCH * SEQ);
    // q2 = facts_q @ Wk^T (bf16 only)
    k_gemm_abT<<<g1, 256, 0, stream>>>(factsq_bf, Wk_bf, nullptr, nullptr, q2_bf, BATCH * SEQ, DIM, DIM);
    // vproj[l,r] = law_embs[l,r,:] . wvd + bv.w_dprob
    k_rowdot<<<(LAWC * LSEQ) / 8, 256, 0, stream>>>(law_embs, wvd, wvd + DIM, vproj, LAWC * LSEQ);
    // fused attention (flash-style, bf16 WMMA, async double-buffered key tiles)
    k_attention<<<BATCH * LAWC * 4, 256, 0, stream>>>(q2_bf, c_fs, lawe_bf, law_masks, vproj, b_dprob, laws2);
    // epilogue
    k_final<<<1, 256, 0, stream>>>(laws2, w_lawprob, b_lawprob, law0, W_rulelaw, b_rulelaw,
                                   W_factlaw, b_factlaw, laws_table, W_law, b_law, d_out);
}
